// RouteGNN_30288109372103
// MI455X (gfx1250) — matrix operand order repaired and checked by reading, not compiled, not run on hardware
//
#include <hip/hip_runtime.h>
#include <hip/hip_bf16.h>
#include <stdint.h>

// ---------------------------------------------------------------------------
// RouteGNN for MI455X (gfx1250): 3x SAGEConv (mean) + edge MLP
// f16 WMMA path + TDM (tensor_load_to_lds) double-buffered weight staging
// ---------------------------------------------------------------------------

#define N_NODES 20000
#define N_EDGES 300000
#define HDIM    256

typedef __attribute__((ext_vector_type(16))) _Float16 v16h;
typedef __attribute__((ext_vector_type(8)))  _Float16 v8h;
typedef __attribute__((ext_vector_type(8)))  float    v8f;
typedef __attribute__((ext_vector_type(4)))  unsigned int u32x4;
typedef __attribute__((ext_vector_type(8)))  int      i32x8;
typedef __attribute__((ext_vector_type(4)))  int      i32x4;

// ---- WMMA fragment helpers (16x16x32 f16, wave32) -------------------------
// A-matrix (16Mx32K) 16-bit layout: lane = M (lane&15); per-lane 16 halves:
//   halves 0..7  -> K = (lane>=16 ? 8 : 0) + 0..7
//   halves 8..15 -> K = 16 + (lane>=16 ? 8 : 0) + 0..7
__device__ __forceinline__ v16h load_a_frag_ptr(const _Float16* rowptr, int lane) {
    const int koff = (lane >> 4) * 8;
    v8h lo = *(const v8h*)(rowptr + koff);
    v8h hi = *(const v8h*)(rowptr + 16 + koff);
    return __builtin_shufflevector(lo, hi, 0,1,2,3,4,5,6,7,8,9,10,11,12,13,14,15);
}

__device__ __forceinline__ v16h load_a_frag(const _Float16* base, int ld,
                                            int row0, int kbase, int lane) {
    const int m = lane & 15;
    return load_a_frag_ptr(base + (size_t)(row0 + m) * ld + kbase, lane);
}

// B fragments pre-packed so each lane does one contiguous 32B load:
// packed[((ntile*ksteps + kstep)*32 + lane)*16 + i] = W[k][n] with
//   k = kstep*32 + (i>>3)*16 + (lane>>4)*8 + (i&7),  n = ntile*16 + (lane&15)
__device__ __forceinline__ v16h load_b_frag(const _Float16* packed, int ksteps,
                                            int ntile, int kstep, int lane) {
    const size_t idx = ((size_t)(ntile * ksteps + kstep) * 32 + lane) * 16;
    return *(const v16h*)(packed + idx);
}

__device__ __forceinline__ v8f wmma_f16(v16h a, v16h b, v8f c) {
    return __builtin_amdgcn_wmma_f32_16x16x32_f16(false, a, false, b,
                                                  (short)0, c, false, false);
}

// ---- Tensor Data Mover: copy one k-step weight slice (16 ntiles x 1KB) ----
// Packed layout: chunk for (ntile, kstep) lives at byte offset
// (ntile*8 + kstep)*1024.  For fixed kstep this is a 2D tile:
//   16 rows x 1KB, row stride 8KB.  data_size = 8B units.
__device__ __forceinline__ void tdm_load_slice(const _Float16* gsrc, unsigned lds_off) {
    unsigned long long ga = (unsigned long long)(uintptr_t)gsrc;
    u32x4 g0;
    g0[0] = 1u;                                                // count=1 (valid D#)
    g0[1] = lds_off;                                           // lds_addr (bytes)
    g0[2] = (unsigned)(ga & 0xFFFFFFFFu);                      // global_addr[31:0]
    g0[3] = (unsigned)((ga >> 32) & 0x01FFFFFFu) | (2u << 30); // addr[56:32] | type=2

    const unsigned td0 = 128;   // tensor_dim0  (8B units) = 1KB row
    const unsigned td1 = 16;    // tensor_dim1  = 16 rows (ntiles)
    const unsigned tl0 = 128;   // tile_dim0
    const unsigned tl1 = 16;    // tile_dim1
    const unsigned st0 = 1024;  // tensor_dim0_stride (8B units) = 8KB
    i32x8 g1;
    g1[0] = (int)(3u << 16);                           // wg_mask=0, data_size=8B
    g1[1] = (int)((td0 & 0xFFFFu) << 16);              // [63:48] tensor_dim0 lo16
    g1[2] = (int)((td0 >> 16) | ((td1 & 0xFFFFu) << 16)); // dim0 hi16 | dim1 lo16
    g1[3] = (int)((td1 >> 16) | (tl0 << 16));          // dim1 hi16 | tile_dim0
    g1[4] = (int)(tl1);                                // tile_dim1 | tile_dim2=0
    g1[5] = (int)(st0);                                // stride0[31:0]
    g1[6] = 0;                                         // stride0[47:32] | stride1 lo
    g1[7] = 0;                                         // stride1 hi
    i32x4 z4 = {0, 0, 0, 0};                           // groups 2/3 unused (2D)
    i32x8 z8 = {0, 0, 0, 0, 0, 0, 0, 0};               // extra group (unused)
    __builtin_amdgcn_tensor_load_to_lds(g0, g1, z4, z4, z8, 0);
}

// ---- Utility kernels ------------------------------------------------------
__global__ void rgnn_zero_f32(float* p, int n) {
    int i = blockIdx.x * 256 + threadIdx.x;
    if (i < n) p[i] = 0.0f;
}

// Pack row-major f32 W[K][N] into f16 fragment-major order (K%32==0, N%16==0).
__global__ void rgnn_pack_b(const float* __restrict__ W, _Float16* __restrict__ P,
                            int K, int N) {
    int tid = blockIdx.x * 256 + threadIdx.x;
    if (tid >= K * N) return;
    int i     = tid & 15;
    int lane  = (tid >> 4) & 31;
    int rest  = tid >> 9;
    int ksteps = K >> 5;
    int kstep = rest % ksteps;
    int ntile = rest / ksteps;
    int k = kstep * 32 + (i >> 3) * 16 + (lane >> 4) * 8 + (i & 7);
    int n = ntile * 16 + (lane & 15);
    P[tid] = (_Float16)W[(size_t)k * N + n];
}

// ---- Layer 1: degree + 3-channel scatter, then tiny dense update ----------
__global__ void rgnn_scatter_x3(const float* __restrict__ x,
                                const int* __restrict__ row,
                                const int* __restrict__ col,
                                float* __restrict__ aggr3,
                                float* __restrict__ cnt) {
    int e = blockIdx.x * 256 + threadIdx.x;
    if (e >= N_EDGES) return;
    int r = row[e], c = col[e];
    atomicAdd(&aggr3[c * 3 + 0], x[r * 3 + 0]);
    atomicAdd(&aggr3[c * 3 + 1], x[r * 3 + 1]);
    atomicAdd(&aggr3[c * 3 + 2], x[r * 3 + 2]);
    atomicAdd(&cnt[c], 1.0f);
}

__global__ void rgnn_layer1(const float* __restrict__ x,
                            const float* __restrict__ aggr3,
                            const float* __restrict__ cnt,
                            const float* __restrict__ Wl,  // [3,256]
                            const float* __restrict__ bl,  // [256]
                            const float* __restrict__ Wr,  // [3,256]
                            float* __restrict__ h32, _Float16* __restrict__ h16) {
    int idx = blockIdx.x * 256 + threadIdx.x;   // n*256 + j
    int n = idx >> 8, j = idx & 255;
    if (n >= N_NODES) return;
    float inv = 1.0f / fmaxf(cnt[n], 1.0f);
    float a0 = aggr3[n * 3 + 0] * inv, a1 = aggr3[n * 3 + 1] * inv, a2 = aggr3[n * 3 + 2] * inv;
    float x0 = x[n * 3 + 0], x1 = x[n * 3 + 1], x2 = x[n * 3 + 2];
    float v = bl[j]
            + a0 * Wl[j] + a1 * Wl[256 + j] + a2 * Wl[512 + j]
            + x0 * Wr[j] + x1 * Wr[256 + j] + x2 * Wr[512 + j];
    v = fmaxf(v, 0.0f);
    h32[idx] = v;
    h16[idx] = (_Float16)v;
}

// ---- 256-wide scatter-add (message aggregation) ---------------------------
__global__ void rgnn_scatter_h(const float* __restrict__ h32,
                               const int* __restrict__ row,
                               const int* __restrict__ col,
                               float* __restrict__ aggr) {
    int e = blockIdx.x;
    int c = threadIdx.x;
    int r = row[e], cl = col[e];
    atomicAdd(&aggr[(size_t)cl * HDIM + c], h32[(size_t)r * HDIM + c]);
}

__global__ void rgnn_mean_div(const float* __restrict__ aggr,
                              const float* __restrict__ cnt,
                              _Float16* __restrict__ aggr16) {
    int idx = blockIdx.x * 256 + threadIdx.x;
    int n = idx >> 8;
    if (n >= N_NODES) return;
    float inv = 1.0f / fmaxf(cnt[n], 1.0f);
    aggr16[idx] = (_Float16)(aggr[idx] * inv);
}

// ---- SAGE dense update: relu(aggr16@Wl + hin16@Wr + bl) -------------------
// M=20000 (32 rows/block), N=K=256.  Weight slices DMA'd to LDS by the TDM,
// double-buffered; 4 waves x (2 Mtiles x 4 Ntiles) each.
__global__ __launch_bounds__(128)
void rgnn_sage_gemm(const _Float16* __restrict__ aggr16,
                    const _Float16* __restrict__ hin16,
                    const _Float16* __restrict__ Wlp,
                    const _Float16* __restrict__ Wrp,
                    const float* __restrict__ bl,
                    float* __restrict__ hout32, _Float16* __restrict__ hout16) {
    __shared__ _Float16 smem[2][2][8192];       // [buf][mat][16 ntiles * 512]
    const int lane = threadIdx.x & 31;
    const int wave = threadIdx.x >> 5;          // 0..3: 4 ntiles each
    const int row0 = blockIdx.x * 32;           // 625 blocks
    const bool leader = (wave == 0);

    if (leader) {
        tdm_load_slice(Wlp, (unsigned)(uintptr_t)&smem[0][0][0]);
        tdm_load_slice(Wrp, (unsigned)(uintptr_t)&smem[0][1][0]);
    }

    v8f acc[2][4] = {};
    int buf = 0;
#pragma unroll 1
    for (int kstep = 0; kstep < 8; ++kstep) {
        if (leader) __builtin_amdgcn_s_wait_tensorcnt(0);
        __syncthreads();                        // slice `buf` ready for all waves
        if (leader && kstep < 7) {              // prefetch next slice into buf^1
            tdm_load_slice(Wlp + (kstep + 1) * 512,
                           (unsigned)(uintptr_t)&smem[buf ^ 1][0][0]);
            tdm_load_slice(Wrp + (kstep + 1) * 512,
                           (unsigned)(uintptr_t)&smem[buf ^ 1][1][0]);
        }
        const int kbase = kstep * 32;
        v16h aA0 = load_a_frag(aggr16, HDIM, row0,      kbase, lane);
        v16h aH0 = load_a_frag(hin16,  HDIM, row0,      kbase, lane);
        v16h aA1 = load_a_frag(aggr16, HDIM, row0 + 16, kbase, lane);
        v16h aH1 = load_a_frag(hin16,  HDIM, row0 + 16, kbase, lane);
#pragma unroll
        for (int t = 0; t < 4; ++t) {
            const int ntile = wave * 4 + t;
            v16h bL = *(const v16h*)&smem[buf][0][(ntile * 32 + lane) * 16];
            v16h bR = *(const v16h*)&smem[buf][1][(ntile * 32 + lane) * 16];
            acc[0][t] = wmma_f16(aA0, bL, acc[0][t]);
            acc[0][t] = wmma_f16(aH0, bR, acc[0][t]);
            acc[1][t] = wmma_f16(aA1, bL, acc[1][t]);
            acc[1][t] = wmma_f16(aH1, bR, acc[1][t]);
        }
        __syncthreads();                        // everyone done with `buf`
        buf ^= 1;
    }

    // C layout: VGPR r, lanes 0-15 -> M=r, lanes 16-31 -> M=8+r; N = lane&15
    const int mb = (lane >> 4) * 8;
    const int nl = lane & 15;
#pragma unroll
    for (int mt = 0; mt < 2; ++mt) {
#pragma unroll
        for (int t = 0; t < 4; ++t) {
            const int n = (wave * 4 + t) * 16 + nl;
            const float bv = bl[n];
#pragma unroll
            for (int r = 0; r < 8; ++r) {
                const float v = fmaxf(acc[mt][t][r] + bv, 0.0f);
                const size_t o = (size_t)(row0 + mt * 16 + mb + r) * HDIM + n;
                hout32[o] = v;
                hout16[o] = (_Float16)v;
            }
        }
    }
}

// ---- Edge MLP layer 1: gather [h[row]|h[col]] (K=512) + scalar epilogue ---
__global__ __launch_bounds__(128)
void rgnn_edge_mlp1(const _Float16* __restrict__ h16,
                    const int* __restrict__ row,
                    const int* __restrict__ col,
                    const float* __restrict__ edge_attr,
                    const float* __restrict__ target_dist,
                    const float* __restrict__ W1,   // [514,256] f32, rows 512/513
                    const float* __restrict__ b1,
                    const _Float16* __restrict__ W1p,  // packed rows 0..511
                    _Float16* __restrict__ z1) {
    const int lane = threadIdx.x & 31;
    const int wave = threadIdx.x >> 5;
    const int e0 = blockIdx.x * 16;             // 18750 blocks
    const int eL = e0 + (lane & 15);
    const _Float16* pR = h16 + (size_t)row[eL] * HDIM;
    const _Float16* pC = h16 + (size_t)col[eL] * HDIM;
    v8f acc[4] = {};
#pragma unroll
    for (int kstep = 0; kstep < 16; ++kstep) {
        const _Float16* src = (kstep < 8 ? pR : pC) + (kstep & 7) * 32;
        v16h a = load_a_frag_ptr(src, lane);
#pragma unroll
        for (int t = 0; t < 4; ++t) {
            v16h b = load_b_frag(W1p, 16, wave * 4 + t, kstep, lane);
            acc[t] = wmma_f16(a, b, acc[t]);
        }
    }
    const float td = target_dist[0];
    const int mb = (lane >> 4) * 8;
    const int nl = lane & 15;
#pragma unroll
    for (int t = 0; t < 4; ++t) {
        const int n = (wave * 4 + t) * 16 + nl;
        const float w512 = W1[512 * 256 + n];
        const float w513 = W1[513 * 256 + n];
        const float bv = b1[n];
#pragma unroll
        for (int r = 0; r < 8; ++r) {
            const int e = e0 + mb + r;
            float v = acc[t][r] + bv + edge_attr[e] * w512 + td * w513;
            z1[(size_t)e * 256 + n] = (_Float16)fmaxf(v, 0.0f);
        }
    }
}

// ---- Edge MLP layers 2+3 fused: (relu(z1@W2+b2))@W3 + b3 ------------------
__global__ __launch_bounds__(128)
void rgnn_edge_mlp23(const _Float16* __restrict__ z1,
                     const _Float16* __restrict__ W2p,  // packed [256,128]
                     const float* __restrict__ b2,
                     const float* __restrict__ W3,      // [128,1] f32
                     const float* __restrict__ b3,
                     float* __restrict__ out) {
    __shared__ float sdot[16];
    const int tid = threadIdx.x;
    const int lane = tid & 31;
    const int wave = tid >> 5;                  // 4 waves x 2 ntiles = N=128
    const int e0 = blockIdx.x * 16;
    if (tid < 16) sdot[tid] = 0.0f;
    __syncthreads();

    v8f acc[2] = {};
#pragma unroll
    for (int kstep = 0; kstep < 8; ++kstep) {
        v16h a = load_a_frag(z1, 256, e0, kstep * 32, lane);
#pragma unroll
        for (int t = 0; t < 2; ++t) {
            v16h b = load_b_frag(W2p, 8, wave * 2 + t, kstep, lane);
            acc[t] = wmma_f16(a, b, acc[t]);
        }
    }
    const int mb = (lane >> 4) * 8;
    const int nl = lane & 15;
    float contrib[8] = {0.f, 0.f, 0.f, 0.f, 0.f, 0.f, 0.f, 0.f};
#pragma unroll
    for (int t = 0; t < 2; ++t) {
        const int n = (wave * 2 + t) * 16 + nl;
        const float bv = b2[n];
        const float w3 = W3[n];
#pragma unroll
        for (int r = 0; r < 8; ++r)
            contrib[r] += fmaxf(acc[t][r] + bv, 0.0f) * w3;
    }
#pragma unroll
    for (int r = 0; r < 8; ++r)
        atomicAdd(&sdot[mb + r], contrib[r]);   // LDS ds_add_f32 reduction
    __syncthreads();
    if (tid < 16) out[e0 + tid] = sdot[tid] + b3[0];
}

// ---------------------------------------------------------------------------
extern "C" void kernel_launch(void* const* d_in, const int* in_sizes, int n_in,
                              void* d_out, int out_size, void* d_ws, size_t ws_size,
                              hipStream_t stream) {
    // Inputs per setup_inputs() order
    const float* x   = (const float*)d_in[0];
    const int*   ei  = (const int*)d_in[1];
    const float* ea  = (const float*)d_in[2];
    const float* td  = (const float*)d_in[3];
    const float* Wl1 = (const float*)d_in[4];
    const float* bl1 = (const float*)d_in[5];
    const float* Wr1 = (const float*)d_in[6];
    const float* Wl2 = (const float*)d_in[7];
    const float* bl2 = (const float*)d_in[8];
    const float* Wr2 = (const float*)d_in[9];
    const float* Wl3 = (const float*)d_in[10];
    const float* bl3 = (const float*)d_in[11];
    const float* Wr3 = (const float*)d_in[12];
    const float* W1  = (const float*)d_in[13];
    const float* b1  = (const float*)d_in[14];
    const float* W2  = (const float*)d_in[15];
    const float* b2  = (const float*)d_in[16];
    const float* W3  = (const float*)d_in[17];
    const float* b3  = (const float*)d_in[18];
    const int* row = ei;
    const int* col = ei + N_EDGES;
    float* out = (float*)d_out;

    // Workspace carve-up (256B aligned)
    char* ws = (char*)d_ws;
    size_t off = 0;
    auto carve = [&](size_t bytes) -> void* {
        void* p = ws + off;
        off = (off + bytes + 255) & ~(size_t)255;
        return p;
    };
    const size_t NH = (size_t)N_NODES * HDIM;
    float*     cnt    = (float*)carve(N_NODES * 4);
    float*     aggr3  = (float*)carve((size_t)N_NODES * 3 * 4);
    float*     hA32   = (float*)carve(NH * 4);
    float*     hB32   = (float*)carve(NH * 4);
    float*     aggr32 = (float*)carve(NH * 4);
    _Float16*  hA16   = (_Float16*)carve(NH * 2);
    _Float16*  hB16   = (_Float16*)carve(NH * 2);
    _Float16*  aggr16 = (_Float16*)carve(NH * 2);
    _Float16*  Wl2p   = (_Float16*)carve(256 * 256 * 2);
    _Float16*  Wr2p   = (_Float16*)carve(256 * 256 * 2);
    _Float16*  Wl3p   = (_Float16*)carve(256 * 256 * 2);
    _Float16*  Wr3p   = (_Float16*)carve(256 * 256 * 2);
    _Float16*  W1p    = (_Float16*)carve(512 * 256 * 2);
    _Float16*  W2p    = (_Float16*)carve(256 * 128 * 2);
    _Float16*  z1     = (_Float16*)carve((size_t)N_EDGES * 256 * 2);
    (void)ws_size; (void)n_in; (void)in_sizes; (void)out_size;

    // 1) Pack weights into WMMA fragment order (f32 -> f16)
    rgnn_pack_b<<<256, 256, 0, stream>>>(Wl2, Wl2p, 256, 256);
    rgnn_pack_b<<<256, 256, 0, stream>>>(Wr2, Wr2p, 256, 256);
    rgnn_pack_b<<<256, 256, 0, stream>>>(Wl3, Wl3p, 256, 256);
    rgnn_pack_b<<<256, 256, 0, stream>>>(Wr3, Wr3p, 256, 256);
    rgnn_pack_b<<<512, 256, 0, stream>>>(W1, W1p, 512, 256);   // rows 0..511
    rgnn_pack_b<<<128, 256, 0, stream>>>(W2, W2p, 256, 128);

    // 2) Degrees + layer 1 (din=3, VALU)
    rgnn_zero_f32<<<(N_NODES + 255) / 256, 256, 0, stream>>>(cnt, N_NODES);
    rgnn_zero_f32<<<(N_NODES * 3 + 255) / 256, 256, 0, stream>>>(aggr3, N_NODES * 3);
    rgnn_scatter_x3<<<(N_EDGES + 255) / 256, 256, 0, stream>>>(x, row, col, aggr3, cnt);
    rgnn_layer1<<<N_NODES, 256, 0, stream>>>(x, aggr3, cnt, Wl1, bl1, Wr1, hA32, hA16);

    // 3) Layer 2: aggregate h1, GEMM (WMMA + TDM-staged weights)
    rgnn_zero_f32<<<(int)(NH / 256), 256, 0, stream>>>(aggr32, (int)NH);
    rgnn_scatter_h<<<N_EDGES, 256, 0, stream>>>(hA32, row, col, aggr32);
    rgnn_mean_div<<<N_NODES, 256, 0, stream>>>(aggr32, cnt, aggr16);
    rgnn_sage_gemm<<<N_NODES / 32, 128, 0, stream>>>(aggr16, hA16, Wl2p, Wr2p, bl2, hB32, hB16);

    // 4) Layer 3
    rgnn_zero_f32<<<(int)(NH / 256), 256, 0, stream>>>(aggr32, (int)NH);
    rgnn_scatter_h<<<N_EDGES, 256, 0, stream>>>(hB32, row, col, aggr32);
    rgnn_mean_div<<<N_NODES, 256, 0, stream>>>(aggr32, cnt, aggr16);
    rgnn_sage_gemm<<<N_NODES / 32, 128, 0, stream>>>(aggr16, hB16, Wl3p, Wr3p, bl3, hA32, hA16);

    // 5) Edge MLP: gather-GEMM (K=512) + scalar-feature epilogue, then fused 2+3
    rgnn_edge_mlp1<<<N_EDGES / 16, 128, 0, stream>>>(hA16, row, col, ea, td, W1, b1, W1p, z1);
    rgnn_edge_mlp23<<<N_EDGES / 16, 128, 0, stream>>>(z1, W2p, b2, W3, b3, out);
}
